// ENAHPool_65223373357284
// MI455X (gfx1250) — matrix operand order repaired and checked by reading, not compiled, not run on hardware
//
#include <hip/hip_runtime.h>
#include <hip/hip_bf16.h>
#include <stdint.h>

typedef _Float16 f16;
typedef __attribute__((ext_vector_type(16))) _Float16 v16h;
typedef __attribute__((ext_vector_type(8)))  float    v8f;

#define N_NODES 8192
#define E_EDGES 262144
#define FEPS 1e-16f

// order-preserving float<->uint mapping for atomicMax on floats
__device__ __forceinline__ unsigned enc_f32(float f) {
  unsigned u = __float_as_uint(f);
  return (u & 0x80000000u) ? ~u : (u | 0x80000000u);
}
__device__ __forceinline__ float dec_f32(unsigned k) {
  unsigned u = (k & 0x80000000u) ? (k & 0x7FFFFFFFu) : ~k;
  return __uint_as_float(u);
}

// ---- CDNA5 async global->LDS copy (ASYNCcnt path, no VGPR staging) ----
// VGLOBAL async encoding: LDS address is taken from the VDST VGPR; generic shared-ptr
// low 32 bits are the group-segment offset, which is exactly what VDST wants.
__device__ __forceinline__ void async_copy_b128(const void* gsrc, void* ldst) {
  unsigned lds = (unsigned)(uintptr_t)ldst;
  unsigned long long ga = (unsigned long long)(uintptr_t)gsrc;
  asm volatile("global_load_async_to_lds_b128 %0, %1, off"
               :: "v"(lds), "v"(ga) : "memory");
}
__device__ __forceinline__ void wait_async0() {
#if __has_builtin(__builtin_amdgcn_s_wait_asynccnt)
  __builtin_amdgcn_s_wait_asynccnt(0);
#else
  asm volatile("s_wait_asynccnt 0x0" ::: "memory");
#endif
}

// ---------------- init ----------------
__global__ void init_kernel(unsigned* __restrict__ mkey, float* __restrict__ ssum,
                            unsigned* __restrict__ cnt, unsigned* __restrict__ bitmap,
                            float* __restrict__ out, int nk, int outlen) {
  int i = blockIdx.x * blockDim.x + threadIdx.x;
  if (i < nk) { mkey[i] = 0x007FFFFFu; /* enc(-inf) */ ssum[i] = 0.f; bitmap[i] = 0u; }
  if (i < N_NODES) cnt[i] = 0u;
  if (i < outlen) out[i] = 0.f;
}

// ---------------- pack 256x256 f32 B into WMMA B-fragment order (f16) ----------------
// flat idx = ((kt*16 + nt)*32 + L)*16 + e ; value = B[kt*32 + (L<16?0:8) + (e&7) + (e&8?16:0)][nt*16 + L%16]
__global__ void pack_b_kernel(const float* __restrict__ B, f16* __restrict__ Bpk) {
  int idx = blockIdx.x * 256 + threadIdx.x;            // 65536 total
  int e  = idx & 15;
  int L  = (idx >> 4) & 31;
  int nt = (idx >> 9) & 15;
  int kt = idx >> 13;
  int k = kt * 32 + ((L < 16) ? 0 : 8) + (e & 7) + ((e & 8) ? 16 : 0);
  int n = nt * 16 + (L & 15);
  Bpk[idx] = (f16)B[k * 256 + n];
}

// ---------------- cvec = be @ Wa + ba ----------------
__global__ void cvec_kernel(const float* __restrict__ be, const float* __restrict__ Wa,
                            const float* __restrict__ ba, float* __restrict__ cvec) {
  int j = threadIdx.x;
  float s = ba[j];
  for (int k = 0; k < 256; ++k) s += be[k] * Wa[k * 256 + j];
  cvec[j] = s;
}

// ---------------- main WMMA GEMM: Out[M x 256] = A[M x 256] @ B + bias ----------------
// mode 0: write Out (+bias). mode 1: att epilogue, atomicMax mkey. mode 2: att epilogue,
//         exp(att - m) atomicAdd into ssum.  (att value never stored)
__global__ __launch_bounds__(256) void gemm_wmma_kernel(
    const float* __restrict__ A, const f16* __restrict__ Bpk,
    const float* __restrict__ bias, float* __restrict__ Out,
    const int* __restrict__ colIdx, const int* __restrict__ rowIdx,
    const float* __restrict__ xa, const float* __restrict__ cvec,
    unsigned* __restrict__ mkey, float* __restrict__ ssum, int mode)
{
  __shared__ f16 Blds[2][8192];                   // double-buffered 32x256 K-chunks
  int tid  = threadIdx.x;
  int lane = tid & 31;
  int wave = tid >> 5;
  size_t m0 = (size_t)blockIdx.x * 128 + (size_t)wave * 16;

  v8f acc[16];
#pragma unroll
  for (int i = 0; i < 16; ++i) { v8f z = {}; acc[i] = z; }

  const float* Arow = A + (m0 + (lane & 15)) * 256;
  const int abase = (lane < 16) ? 0 : 8;

  // prologue: async-stage chunk 0 (16KB, 4 x b128 per thread)
  {
    const f16* src = Bpk;
    f16* dst = &Blds[0][0];
#pragma unroll
    for (int i = 0; i < 4; ++i)
      async_copy_b128(src + (tid + 256 * i) * 8, dst + (tid + 256 * i) * 8);
  }
  wait_async0();
  __syncthreads();

  for (int kt = 0; kt < 8; ++kt) {
    const f16* cur = &Blds[kt & 1][0];
    // async-stage next chunk into the other buffer while we compute
    if (kt + 1 < 8) {
      const f16* src = Bpk + (size_t)(kt + 1) * 8192;
      f16* dst = &Blds[(kt + 1) & 1][0];
#pragma unroll
      for (int i = 0; i < 4; ++i)
        async_copy_b128(src + (tid + 256 * i) * 8, dst + (tid + 256 * i) * 8);
    }

    // A fragment: 16x32, per-lane K = {base..base+7, base+16..base+23}
    v16h afrag;
    const float* ap = Arow + kt * 32 + abase;
#pragma unroll
    for (int e = 0; e < 8; ++e) afrag[e] = (f16)ap[e];
#pragma unroll
    for (int e = 0; e < 8; ++e) afrag[8 + e] = (f16)ap[16 + e];

#pragma unroll
    for (int nt = 0; nt < 16; ++nt) {
      v16h bfrag = *(const v16h*)(cur + (nt * 32 + lane) * 16);
      acc[nt] = __builtin_amdgcn_wmma_f32_16x16x32_f16(
          false, afrag, false, bfrag, (short)0, acc[nt], false, false);
    }

    wait_async0();
    __syncthreads();
  }

  // epilogue (C layout: VGPR v -> M = v + (lane<16?0:8), N = lane%16)
  const int nb = lane & 15;
  const size_t mb = m0 + ((lane < 16) ? 0 : 8);
  if (mode == 0) {
#pragma unroll
    for (int v = 0; v < 8; ++v) {
      size_t m = mb + v;
#pragma unroll
      for (int nt = 0; nt < 16; ++nt) {
        int n = nt * 16 + nb;
        float val = acc[nt][v];
        if (bias) val += bias[n];
        Out[m * 256 + n] = val;
      }
    }
  } else {
#pragma unroll
    for (int v = 0; v < 8; ++v) {
      size_t m = mb + v;
      int c = colIdx[m];
      int r = rowIdx[m];
#pragma unroll
      for (int nt = 0; nt < 16; ++nt) {
        int n = nt * 16 + nb;
        float val = acc[nt][v] + xa[(size_t)c * 256 + n] + xa[(size_t)r * 256 + n] + cvec[n];
        if (mode == 1) {
          atomicMax(&mkey[(size_t)c * 256 + n], enc_f32(val));
        } else {
          float mv = dec_f32(mkey[(size_t)c * 256 + n]);
          atomicAdd(&ssum[(size_t)c * 256 + n], __expf(val - mv));
        }
      }
    }
  }
}

// ---------------- edges: dedup bitmap (adj uses .set semantics) + segment counts ----------------
__global__ void edge_kernel(const int* __restrict__ rowIdx, const int* __restrict__ colIdx,
                            unsigned* __restrict__ bitmap, unsigned* __restrict__ cnt) {
  size_t e = (size_t)blockIdx.x * blockDim.x + threadIdx.x;
  if (e >= E_EDGES) return;
  int r = rowIdx[e], c = colIdx[e];
  atomicOr(&bitmap[(size_t)r * 256 + (c >> 5)], 1u << (c & 31));
  atomicAdd(&cnt[c], 1u);
}

// ---------------- S = softmax( (ssum/(ssum+eps)) / max(cnt,1) , axis=-1 ) ----------------
__global__ __launch_bounds__(256) void softmax_kernel(const float* __restrict__ ssum,
                                                      const unsigned* __restrict__ cnt,
                                                      float* __restrict__ S) {
  __shared__ float red[256];
  int n = blockIdx.x, k = threadIdx.x;
  unsigned cu = cnt[n];
  float cn = (float)(cu ? cu : 1u);
  float sv = ssum[(size_t)n * 256 + k];
  float a = (sv / (sv + FEPS)) / cn;         // == segsum(att_sm)/max(cnt,1) exactly
  red[k] = a; __syncthreads();
  for (int s = 128; s > 0; s >>= 1) { if (k < s) red[k] = fmaxf(red[k], red[k + s]); __syncthreads(); }
  float mx = red[0]; __syncthreads();
  float ex = __expf(a - mx);
  red[k] = ex; __syncthreads();
  for (int s = 128; s > 0; s >>= 1) { if (k < s) red[k] += red[k + s]; __syncthreads(); }
  S[(size_t)n * 256 + k] = ex / red[0];
}

// ---------------- T = adj @ S via dedup bitmap ----------------
__global__ __launch_bounds__(256) void adjs_kernel(const unsigned* __restrict__ bitmap,
                                                   const float* __restrict__ S,
                                                   float* __restrict__ T) {
  __shared__ unsigned words[256];
  int n = blockIdx.x, t = threadIdx.x;
  words[t] = bitmap[(size_t)n * 256 + t];
  __syncthreads();
  float acc = 0.f;
  for (int w = 0; w < 256; ++w) {
    unsigned b = words[w];
    while (b) {
      int c = (w << 5) + __ffs(b) - 1;
      b &= b - 1;
      acc += S[(size_t)c * 256 + t];
    }
  }
  T[(size_t)n * 256 + t] = acc;
}

// ---------------- Out[256x256] += A[8192x256]^T @ B[8192x256] (split-K, WMMA) ----------------
__global__ __launch_bounds__(256) void atb_wmma_kernel(const float* __restrict__ A,
                                                       const float* __restrict__ B,
                                                       float* __restrict__ Out) {
  __shared__ f16 Bl[8192];   // 32x256 B chunk, B-fragment packed
  __shared__ f16 Al[4096];   // 32x128 A^T chunk, A-fragment packed
  int tid = threadIdx.x, lane = tid & 31, wave = tid >> 5;
  int ibase = blockIdx.y * 128;              // output-row (cluster) group
  int k0b = blockIdx.x * 128;                // K chunk

  v8f acc[16];
#pragma unroll
  for (int i = 0; i < 16; ++i) { v8f z = {}; acc[i] = z; }

  for (int kt = 0; kt < 4; ++kt) {
    int k0 = k0b + kt * 32;
    int kk = tid >> 3;                       // 0..31: K within chunk
    // B chunk: thread reads 32 contiguous floats, scatters into B-frag layout
    {
      int n0 = (tid & 7) * 32;
      const float* bp = B + (size_t)(k0 + kk) * 256 + n0;
      int Lk = (kk >= 16) ? 16 : 0;
      int e = kk & 15;
#pragma unroll
      for (int i = 0; i < 32; ++i) {
        int n = n0 + i;
        Bl[(((n >> 4) * 32) + (n & 15) + Lk) * 16 + e] = (f16)bp[i];
      }
    }
    // A^T chunk: thread reads 16 contiguous floats (one row slice), scatters into A-frag layout
    {
      int c0 = (tid & 7) * 16;
      const float* apg = A + (size_t)(k0 + kk) * 256 + ibase + c0;
      int Lk = (kk >= 16) ? 16 : 0;
      int e = kk & 15;
#pragma unroll
      for (int i = 0; i < 16; ++i) {
        int ii = c0 + i;                     // local output-row 0..127
        Al[(((ii >> 4) * 32) + (ii & 15) + Lk) * 16 + e] = (f16)apg[i];
      }
    }
    __syncthreads();

    v16h af = *(const v16h*)(Al + ((wave * 32) + lane) * 16);
#pragma unroll
    for (int nt = 0; nt < 16; ++nt) {
      v16h bf = *(const v16h*)(Bl + ((nt * 32) + lane) * 16);
      acc[nt] = __builtin_amdgcn_wmma_f32_16x16x32_f16(
          false, af, false, bf, (short)0, acc[nt], false, false);
    }
    __syncthreads();
  }

  int ib = ibase + wave * 16 + ((lane < 16) ? 0 : 8);
  int nb = lane & 15;
#pragma unroll
  for (int v = 0; v < 8; ++v) {
#pragma unroll
    for (int nt = 0; nt < 16; ++nt) {
      atomicAdd(&Out[(size_t)(ib + v) * 256 + nt * 16 + nb], acc[nt][v]);
    }
  }
}

extern "C" void kernel_launch(void* const* d_in, const int* in_sizes, int n_in,
                              void* d_out, int out_size, void* d_ws, size_t ws_size,
                              hipStream_t stream) {
  const float* x     = (const float*)d_in[0];
  const int*   eidx  = (const int*)d_in[1];
  const float* eattr = (const float*)d_in[2];
  const float* Wn    = (const float*)d_in[3];
  const float* bn    = (const float*)d_in[4];
  const float* We    = (const float*)d_in[5];
  const float* be    = (const float*)d_in[6];
  const float* Wa    = (const float*)d_in[7];
  const float* ba    = (const float*)d_in[8];
  const int* rowIdx = eidx;
  const int* colIdx = eidx + E_EDGES;
  float* out = (float*)d_out;

  char* p = (char*)d_ws;
  auto alloc = [&](size_t bytes) { void* r = (void*)p; p += (bytes + 255) & ~(size_t)255; return r; };
  float*    x_proj = (float*)alloc(sizeof(float) * N_NODES * 256);
  float*    xa     = (float*)alloc(sizeof(float) * N_NODES * 256);
  float*    Wea    = (float*)alloc(sizeof(float) * 65536);
  float*    cvec   = (float*)alloc(sizeof(float) * 256);
  f16*      Wn_pk  = (f16*)alloc(sizeof(f16) * 65536);
  f16*      Wa_pk  = (f16*)alloc(sizeof(f16) * 65536);
  f16*      Wea_pk = (f16*)alloc(sizeof(f16) * 65536);
  unsigned* mkey   = (unsigned*)alloc(4ull * N_NODES * 256);
  float*    ssum   = (float*)alloc(4ull * N_NODES * 256);
  unsigned* cnt    = (unsigned*)alloc(4ull * N_NODES);
  float*    Smat   = (float*)alloc(4ull * N_NODES * 256);
  unsigned* bitmap = (unsigned*)alloc(4ull * N_NODES * 256);
  float*    Tmat   = (float*)alloc(4ull * N_NODES * 256);

  init_kernel<<<N_NODES, 256, 0, stream>>>(mkey, ssum, cnt, bitmap, out, N_NODES * 256, 131072);
  edge_kernel<<<E_EDGES / 256, 256, 0, stream>>>(rowIdx, colIdx, bitmap, cnt);
  pack_b_kernel<<<256, 256, 0, stream>>>(Wn, Wn_pk);
  pack_b_kernel<<<256, 256, 0, stream>>>(Wa, Wa_pk);
  cvec_kernel<<<1, 256, 0, stream>>>(be, Wa, ba, cvec);

  // x_proj = x @ Wn + bn
  gemm_wmma_kernel<<<N_NODES / 128, 256, 0, stream>>>(x, Wn_pk, bn, x_proj,
      nullptr, nullptr, nullptr, nullptr, nullptr, nullptr, 0);
  // Wea = We @ Wa
  gemm_wmma_kernel<<<2, 256, 0, stream>>>(We, Wa_pk, nullptr, Wea,
      nullptr, nullptr, nullptr, nullptr, nullptr, nullptr, 0);
  pack_b_kernel<<<256, 256, 0, stream>>>(Wea, Wea_pk);
  // xa = x_proj @ Wa
  gemm_wmma_kernel<<<N_NODES / 128, 256, 0, stream>>>(x_proj, Wa_pk, nullptr, xa,
      nullptr, nullptr, nullptr, nullptr, nullptr, nullptr, 0);

  // pass 1: segment max of att = eattr@Wea + xa[col] + xa[row] + cvec
  gemm_wmma_kernel<<<E_EDGES / 128, 256, 0, stream>>>(eattr, Wea_pk, nullptr, nullptr,
      colIdx, rowIdx, xa, cvec, mkey, ssum, 1);
  // pass 2: segment sum of exp(att - m)
  gemm_wmma_kernel<<<E_EDGES / 128, 256, 0, stream>>>(eattr, Wea_pk, nullptr, nullptr,
      colIdx, rowIdx, xa, cvec, mkey, ssum, 2);

  softmax_kernel<<<N_NODES, 256, 0, stream>>>(ssum, cnt, Smat);
  adjs_kernel<<<N_NODES, 256, 0, stream>>>(bitmap, Smat, Tmat);

  // x_pooled = S^T @ x ; adj_pooled = S^T @ (adj @ S)
  atb_wmma_kernel<<<dim3(N_NODES / 128, 2), 256, 0, stream>>>(Smat, x, out);
  atb_wmma_kernel<<<dim3(N_NODES / 128, 2), 256, 0, stream>>>(Smat, Tmat, out + 65536);
}
